// FFTPatchEmbed_68728066671242
// MI455X (gfx1250) — matrix operand order repaired and checked by reading, not compiled, not run on hardware
//
#include <hip/hip_runtime.h>
#include <hip/hip_bf16.h>

// ---------------------------------------------------------------------------
// FFTPatchEmbed for MI455X (gfx1250, wave32, WMMA + async global->LDS DMA).
// P=16, PF=9, B=32, C=3, H=512 -> nP=32, M=32768 rows, K=432 (pad 448), N=768.
// Pipeline: init -> per-patch rFFT2 features (bf16) + global max ->
//           -inf fixup pass -> double-buffered bf16 WMMA GEMM with
//           GLOBAL_LOAD_ASYNC_TO_LDS_B128 prefetch.
// ---------------------------------------------------------------------------

#define NP        32
#define PATCHES   98304        // B*C*nP*nP
#define MROWS     32768        // B*nP*nP
#define KDIM      432
#define KPAD      448
#define NDIM      768
#define KSTEPS    (KPAD / 32)  // 14
#define LSTRIDE   40           // LDS tile stride in halves: 80B rows -> 16B-aligned
                               // async dsts, 20-dword bank stride (conflict-free)

typedef __attribute__((ext_vector_type(16))) __bf16        v16bf;
typedef __attribute__((ext_vector_type(8)))  float         v8f;
typedef __attribute__((ext_vector_type(8)))  unsigned int  v8u;
typedef int v4i_t __attribute__((vector_size(16)));        // matches builtin sig

__device__ __forceinline__ unsigned short f2bf(float f) {
  unsigned u = __float_as_uint(f);
  unsigned r = (u + 0x7FFFu + ((u >> 16) & 1u)) >> 16;   // RNE; -inf stays 0xFF80
  return (unsigned short)r;
}
__device__ __forceinline__ unsigned encodeF(float f) {   // monotone float->uint
  unsigned u = __float_as_uint(f);
  return ((int)u >= 0) ? (u | 0x80000000u) : ~u;
}
__device__ __forceinline__ float decodeF(unsigned k) {
  unsigned u = (k & 0x80000000u) ? (k ^ 0x80000000u) : ~k;
  return __uint_as_float(u);
}

// ---- gfx1250 async global->LDS copy (ASYNCcnt) with inline-asm fallback ----
__device__ __forceinline__ void async_cp_b128(const void* g, void* l) {
#if __has_builtin(__builtin_amdgcn_global_load_async_to_lds_b128)
  __builtin_amdgcn_global_load_async_to_lds_b128(
      (__attribute__((address_space(1))) v4i_t*)(void*)g,
      (__attribute__((address_space(3))) v4i_t*)l, 0, 0);
#else
  // Generic LDS pointer low 32 bits == LDS byte offset (aperture rule).
  unsigned lds = (unsigned)(unsigned long long)l;
  asm volatile("global_load_async_to_lds_b128 %0, %1, off"
               :: "v"(lds), "v"((unsigned long long)g) : "memory");
#endif
}
__device__ __forceinline__ void wait_async0() {
#if __has_builtin(__builtin_amdgcn_s_wait_asynccnt)
  __builtin_amdgcn_s_wait_asynccnt(0);
#else
  asm volatile("s_wait_asynccnt 0x0" ::: "memory");
#endif
}

// ---------------------------------------------------------------------------
// Kernel 0: init gmax slot, convert W (768x432 f32) -> bf16 padded to 768x448,
// and zero feature pad columns K=432..447 for all 32768 rows.
// ---------------------------------------------------------------------------
__global__ void fpe_init_kernel(const float* __restrict__ W,
                                unsigned short* __restrict__ Wbf,
                                unsigned short* __restrict__ feat,
                                unsigned* __restrict__ gmax_u) {
  int idx = blockIdx.x * 256 + threadIdx.x;
  if (idx == 0) *gmax_u = 0x007FFFFFu;            // encodeF(-inf)
  const int NW = NDIM * KPAD;                     // 344064
  if (idx < NW) {
    int n = idx / KPAD, kk = idx - n * KPAD;
    Wbf[idx] = (kk < KDIM) ? f2bf(W[n * KDIM + kk]) : (unsigned short)0;
  }
  int j = idx - NW;
  if (j >= 0 && j < MROWS * (KPAD - KDIM)) {
    int row = j >> 4, kk = KDIM + (j & 15);
    feat[row * KPAD + kk] = 0;
  }
}

// ---------------------------------------------------------------------------
// Kernel 1: per-patch 16x16 rFFT2 (forward norm), log-magnitude -> bf16
// features [row=b*1024+pi*32+pj][k=c*144+ku*9+kv]; global max via atomicMax.
// One patch per wave32, 8 waves per block. Separable DFT through LDS.
// ---------------------------------------------------------------------------
__global__ __launch_bounds__(256) void fpe_fft_kernel(
    const float* __restrict__ x,
    unsigned short* __restrict__ feat,
    unsigned* __restrict__ gmax_u) {
  __shared__ float ct[16], st[16];
  __shared__ float xs[8][16][16];
  __shared__ float Rr[8][16][9];
  __shared__ float Ri[8][16][9];

  const int tid = threadIdx.x;
  const int w   = tid >> 5;
  const int l   = tid & 31;

  if (tid < 16) {
    float s, c;
    __sincosf((float)tid * 0.39269908169872414f, &s, &c);  // 2*pi*m/16
    ct[tid] = c; st[tid] = s;
  }

  const int p    = blockIdx.x * 8 + w;           // patch id
  const int cell = p & 1023;                     // pi*32+pj
  const int bc   = p >> 10;
  const int c    = bc % 3;
  const int b    = bc / 3;
  const int pi   = cell >> 5;
  const int pj   = cell & 31;

  const int r  = l & 15;
  const int hf = l >> 4;

  // load 16x16 patch: lane = (row r, col-half hf) -> 8 contiguous floats
  const float* src = x + (((size_t)(b * 3 + c) * 512 + pi * 16 + r) * 512
                          + pj * 16 + hf * 8);
  float4 v0 = *(const float4*)(src);
  float4 v1 = *(const float4*)(src + 4);
  int cb = hf * 8;
  xs[w][r][cb + 0] = v0.x; xs[w][r][cb + 1] = v0.y;
  xs[w][r][cb + 2] = v0.z; xs[w][r][cb + 3] = v0.w;
  xs[w][r][cb + 4] = v1.x; xs[w][r][cb + 5] = v1.y;
  xs[w][r][cb + 6] = v1.z; xs[w][r][cb + 7] = v1.w;
  __syncthreads();

  // stage 1: row DFTs, real -> 9 complex bins. lane handles row r, kv subset.
  const int kv0 = hf ? 5 : 0;
  const int kv1 = hf ? 9 : 5;
  for (int kv = kv0; kv < kv1; ++kv) {
    float sr = 0.f, si = 0.f;
    #pragma unroll
    for (int n = 0; n < 16; ++n) {
      float xv = xs[w][r][n];
      int m = (kv * n) & 15;
      sr += xv * ct[m];
      si -= xv * st[m];
    }
    Rr[w][r][kv] = sr;
    Ri[w][r][kv] = si;
  }
  __syncthreads();

  // stage 2: column DFTs (ku = r), magnitude*(1/256), log.
  unsigned short* frow =
      feat + (size_t)(b * 1024 + cell) * KPAD + c * 144 + r * 9;
  float vmax = -__builtin_inff();
  for (int kv = kv0; kv < kv1; ++kv) {
    float fr = 0.f, fi = 0.f;
    #pragma unroll
    for (int rr = 0; rr < 16; ++rr) {
      int m = (r * rr) & 15;
      float ar = Rr[w][rr][kv], ai = Ri[w][rr][kv];
      fr += ar * ct[m] + ai * st[m];
      fi += ai * ct[m] - ar * st[m];
    }
    float mag = sqrtf(fr * fr + fi * fi) * (1.0f / 256.0f);
    float val = logf(mag);                       // log(0) -> -inf (fixed later)
    vmax = fmaxf(vmax, val);
    frow[kv] = f2bf(val);
  }

  // wave32 max reduce -> one atomic per patch
  #pragma unroll
  for (int m = 16; m >= 1; m >>= 1)
    vmax = fmaxf(vmax, __shfl_xor(vmax, m, 32));
  if (l == 0) atomicMax(gmax_u, encodeF(vmax));
}

// ---------------------------------------------------------------------------
// Kernel 2: replace bf16 -inf (0xFF80) feature values with the global max.
// Read-mostly pass; stores only dwords that actually change (≈ none).
// ---------------------------------------------------------------------------
__global__ __launch_bounds__(256) void fpe_fixup_kernel(
    unsigned* __restrict__ feat_u,
    const unsigned* __restrict__ gmax_u) {
  const unsigned short g16 = f2bf(decodeF(*gmax_u));
  size_t i = (size_t)blockIdx.x * 256 + threadIdx.x;  // MROWS*KPAD/2 dwords
  unsigned u = feat_u[i];
  unsigned v = u;
  if ((v & 0xFFFFu) == 0xFF80u) v = (v & 0xFFFF0000u) | (unsigned)g16;
  if ((v >> 16)    == 0xFF80u) v = (v & 0x0000FFFFu) | ((unsigned)g16 << 16);
  if (v != u) feat_u[i] = v;
}

// ---------------------------------------------------------------------------
// Kernel 3: bf16 WMMA GEMM, double-buffered via async global->LDS DMA.
// out[b,e,pi,pj] = |feat_row . W[e,:] + bias[e]|
// Block: 128(M) x 64(N), 8 waves, each wave 32x32 (2x2 WMMA 16x16x32 bf16).
// ---------------------------------------------------------------------------
__global__ __launch_bounds__(256) void fpe_gemm_kernel(
    const unsigned short* __restrict__ feat,
    const unsigned short* __restrict__ Wbf,
    const float* __restrict__ bias,
    float* __restrict__ out) {
  __shared__ unsigned short As[2][128 * LSTRIDE];
  __shared__ unsigned short Bs[2][64 * LSTRIDE];

  const int tid = threadIdx.x;
  const int l   = tid & 31;
  const int w   = tid >> 5;
  const int wm  = w & 3;          // wave row 0..3  (32 rows each)
  const int wn  = w >> 2;         // wave col 0..1  (32 cols each)
  const int mb  = blockIdx.y * 128;
  const int nb  = blockIdx.x * 64;

  // async-copy thread mapping: 16B (8-half) chunks
  const int ar  = tid >> 2;            // A rows 0..63 (and +64)
  const int ao  = (tid & 3) * 8;       // half offset within 32-half row chunk
  const int bn  = tid >> 2;            // B row (n) 0..63
  const int bo  = (tid & 3) * 8;

  const unsigned short* ag0 = feat + (size_t)(mb + ar) * KPAD + ao;
  const unsigned short* ag1 = feat + (size_t)(mb + ar + 64) * KPAD + ao;
  const unsigned short* bg  = Wbf  + (size_t)(nb + bn) * KPAD + bo;

  v8f acc[2][2];
  #pragma unroll
  for (int mi = 0; mi < 2; ++mi)
    #pragma unroll
    for (int ni = 0; ni < 2; ++ni)
      acc[mi][ni] = (v8f){0.f, 0.f, 0.f, 0.f, 0.f, 0.f, 0.f, 0.f};

  const int hf = l >> 4, lm = l & 15;

  // prefetch k-step 0 into buffer 0
  async_cp_b128(ag0, &As[0][ar * LSTRIDE + ao]);
  async_cp_b128(ag1, &As[0][(ar + 64) * LSTRIDE + ao]);
  async_cp_b128(bg,  &Bs[0][bn * LSTRIDE + bo]);

  for (int kt = 0; kt < KSTEPS; ++kt) {
    const int cur = kt & 1;
    wait_async0();        // this wave's DMA into buf `cur` complete
    __syncthreads();      // everyone's DMA complete -> tile `cur` valid

    if (kt + 1 < KSTEPS) {
      const int nxt = cur ^ 1;
      const int ko  = (kt + 1) * 32;
      async_cp_b128(ag0 + ko, &As[nxt][ar * LSTRIDE + ao]);
      async_cp_b128(ag1 + ko, &As[nxt][(ar + 64) * LSTRIDE + ao]);
      async_cp_b128(bg  + ko, &Bs[nxt][bn * LSTRIDE + bo]);
    }

    // ---- fragments per ISA 7.12.2 layouts ----
    v16bf af[2], bfr[2];
    #pragma unroll
    for (int mi = 0; mi < 2; ++mi) {
      const unsigned short* ab = &As[cur][(wm * 32 + mi * 16 + lm) * LSTRIDE];
      v8u au;
      #pragma unroll
      for (int j = 0; j < 8; ++j) {
        int K = (j < 4) ? (hf * 8 + j * 2) : (16 + hf * 8 + (j - 4) * 2);
        au[j] = *(const unsigned*)(ab + K);
      }
      af[mi] = __builtin_bit_cast(v16bf, au);
    }
    #pragma unroll
    for (int ni = 0; ni < 2; ++ni) {
      const unsigned short* bb =
          &Bs[cur][(wn * 32 + ni * 16 + lm) * LSTRIDE + hf * 8];
      v8u bu;
      #pragma unroll
      for (int j = 0; j < 8; ++j)
        bu[j] = *(const unsigned*)(bb + ((j < 4) ? j * 2 : 16 + (j - 4) * 2));
      bfr[ni] = __builtin_bit_cast(v16bf, bu);
    }

    #pragma unroll
    for (int mi = 0; mi < 2; ++mi)
      #pragma unroll
      for (int ni = 0; ni < 2; ++ni)
        acc[mi][ni] = __builtin_amdgcn_wmma_f32_16x16x32_bf16(
            false, af[mi], false, bfr[ni], (short)0, acc[mi][ni], false, false);
  }

  // ---- epilogue: +bias, abs, write out[b, e, pi, pj] ----
  const int bimg  = mb >> 10;        // 128 | 1024 -> constant per block
  const int rbase = mb & 1023;
  #pragma unroll
  for (int mi = 0; mi < 2; ++mi) {
    int rloc = wm * 32 + mi * 16 + hf * 8;
    #pragma unroll
    for (int ni = 0; ni < 2; ++ni) {
      int e = nb + wn * 32 + ni * 16 + lm;
      float bv = bias[e];
      float* o = out + ((size_t)bimg * NDIM + e) * 1024 + rbase + rloc;
      v8f a = acc[mi][ni];
      float4 o0 = make_float4(fabsf(a[0] + bv), fabsf(a[1] + bv),
                              fabsf(a[2] + bv), fabsf(a[3] + bv));
      float4 o1 = make_float4(fabsf(a[4] + bv), fabsf(a[5] + bv),
                              fabsf(a[6] + bv), fabsf(a[7] + bv));
      *(float4*)(o)     = o0;
      *(float4*)(o + 4) = o1;
    }
  }
}

// ---------------------------------------------------------------------------
extern "C" void kernel_launch(void* const* d_in, const int* in_sizes, int n_in,
                              void* d_out, int out_size, void* d_ws, size_t ws_size,
                              hipStream_t stream) {
  const float* x = (const float*)d_in[0];   // (32,3,512,512)
  const float* W = (const float*)d_in[1];   // (768,432)
  const float* b = (const float*)d_in[2];   // (768,)
  float* out = (float*)d_out;               // (32,768,32,32)

  char* ws = (char*)d_ws;
  unsigned*       gmax = (unsigned*)ws;                               // 4 B
  unsigned short* Wbf  = (unsigned short*)(ws + 256);                 // 768*448*2
  unsigned short* feat = (unsigned short*)(ws + 256 + NDIM * KPAD * 2);
  // feat: 32768*448*2 = ~28 MB; total ws usage ~29 MB.

  {
    int total = 1 + NDIM * KPAD + MROWS * (KPAD - KDIM);
    int blocks = (total + 255) / 256;
    fpe_init_kernel<<<blocks, 256, 0, stream>>>(W, Wbf, feat, gmax);
  }
  fpe_fft_kernel<<<PATCHES / 8, 256, 0, stream>>>(x, feat, gmax);
  {
    int dwords = MROWS * KPAD / 2;              // 7,340,032
    fpe_fixup_kernel<<<dwords / 256, 256, 0, stream>>>((unsigned*)feat, gmax);
  }
  {
    dim3 grid(NDIM / 64, MROWS / 128);          // (12, 256)
    fpe_gemm_kernel<<<grid, 256, 0, stream>>>(feat, Wbf, b, out);
  }
}
